// Attention_12343736009352
// MI455X (gfx1250) — matrix operand order repaired and checked by reading, not compiled, not run on hardware
//
#include <hip/hip_runtime.h>

typedef __attribute__((ext_vector_type(16))) _Float16 v16h;
typedef __attribute__((ext_vector_type(8)))  _Float16 v8h;
typedef __attribute__((ext_vector_type(8)))  float    v8f;

#define ATT_B 2
#define ATT_H 16
#define ATT_S 2048
#define ATT_D 64
#define BM    128           // query rows per block (16 per wave * 8 waves)
#define BN    64            // K/V columns per iteration
#define KS    72            // Kld row stride in halves (144B, 16B-aligned, bank-spread)
#define VS    72            // VT  row stride in halves
#define PS    64            // Pld row stride in halves (one 16x64 P tile per wave)

// load 16 contiguous f16 from LDS as two 16B-aligned chunks
__device__ inline v16h ld_frag(const _Float16* p) {
    v8h lo = *(const v8h*)p;
    v8h hi = *(const v8h*)(p + 8);
    v16h r;
    #pragma unroll
    for (int i = 0; i < 8; ++i) { r[i] = lo[i]; r[i + 8] = hi[i]; }
    return r;
}

#define WMMA_F16(A, B, C) \
    __builtin_amdgcn_wmma_f32_16x16x32_f16(false, (A), false, (B), (short)0, (C), false, false)

__global__ __launch_bounds__(256, 2)
void fa2_causal_wmma_kernel(const float* __restrict__ Q,
                            const float* __restrict__ K,
                            const float* __restrict__ V,
                            float* __restrict__ O)
{
    __shared__ __align__(16) _Float16 Kld[BN * KS];        // K tile, f16 row-major [k][d]
    __shared__ __align__(16) _Float16 VT [ATT_D * VS];     // V tile, f16 transposed [d][k]
    __shared__ __align__(16) _Float16 Pld[8][16 * PS];     // per-wave P relayout buffer

    const int tid  = threadIdx.x;
    const int lane = tid & 31;
    const int half = lane >> 4;
    const int n    = lane & 15;
    // wave id as a true scalar so causal branches become s_cbranch (wave-uniform)
    const int w    = __builtin_amdgcn_readfirstlane(tid >> 5);

    const int bh = blockIdx.y;
    const size_t headoff = (size_t)bh * ATT_S * ATT_D;
    const float* Qh = Q + headoff;
    const float* Kh = K + headoff;
    const float* Vh = V + headoff;
    float*       Oh = O + headoff;

    const int qlo  = blockIdx.x * BM + w * 16;             // this wave's 16 query rows (sgpr)
    const int qhi  = qlo + 15;
    const int kmax = blockIdx.x * BM + BM - 1;             // block-uniform causal limit

    // staging decomposition (256 threads, 64x64 tiles, 16 elements each)
    const int krow  = tid >> 2;             // 0..63
    const int kcol0 = (tid & 3) << 4;       // 0,16,32,48
    const int vrow  = (tid >> 3) << 1;      // 0,2,..,62 (2 rows)
    const int vcol0 = (tid & 7) << 3;       // 0..56 (8 cols)

    // ---- Q A-fragments, loaded once, scale 1/sqrt(64) folded in ----
    // A fragment, contiguous k order used here (B matches): k = 16*half + 2*j
    v16h aQ0, aQ1;
    {
        const float* qrow = Qh + (size_t)(qlo + n) * ATT_D;
        #pragma unroll
        for (int j = 0; j < 8; ++j) {
            int d = half * 16 + (j << 1);
            float2 p0 = *reinterpret_cast<const float2*>(qrow + d);
            float2 p1 = *reinterpret_cast<const float2*>(qrow + 32 + d);
            aQ0[2*j]   = (_Float16)(p0.x * 0.125f);
            aQ0[2*j+1] = (_Float16)(p0.y * 0.125f);
            aQ1[2*j]   = (_Float16)(p1.x * 0.125f);
            aQ1[2*j+1] = (_Float16)(p1.y * 0.125f);
        }
    }

    // all-ones B fragment: row-sum of P via WMMA (every output column = row sum)
    v16h ones;
    #pragma unroll
    for (int i = 0; i < 16; ++i) ones[i] = (_Float16)1.0f;

    v8f oAcc[4];
    #pragma unroll
    for (int t = 0; t < 4; ++t) oAcc[t] = (v8f){};

    float mrun[8], lrun[8];
    #pragma unroll
    for (int r = 0; r < 8; ++r) { mrun[r] = -1e30f; lrun[r] = 0.0f; }

    for (int kt = 0; kt <= kmax; kt += BN) {               // uniform per BLOCK
        __syncthreads();                                   // WAR: prior reads done

        // ============ cooperative stage: K tile (64x64) -> LDS f16 row-major ============
        {
            const float* src = Kh + (size_t)(kt + krow) * ATT_D + kcol0;
            #pragma unroll
            for (int hc = 0; hc < 2; ++hc) {
                float4 a = *(const float4*)(src + hc * 8);
                float4 b = *(const float4*)(src + hc * 8 + 4);
                v8h h;
                h[0] = (_Float16)a.x; h[1] = (_Float16)a.y;
                h[2] = (_Float16)a.z; h[3] = (_Float16)a.w;
                h[4] = (_Float16)b.x; h[5] = (_Float16)b.y;
                h[6] = (_Float16)b.z; h[7] = (_Float16)b.w;
                *(v8h*)&Kld[krow * KS + kcol0 + hc * 8] = h;
            }
        }
        // ============ cooperative stage: V tile (64x64) -> LDS f16 TRANSPOSED ============
        {
            const float* s0 = Vh + (size_t)(kt + vrow) * ATT_D + vcol0;
            float4 r0a = *(const float4*)s0;
            float4 r0b = *(const float4*)(s0 + 4);
            float4 r1a = *(const float4*)(s0 + ATT_D);
            float4 r1b = *(const float4*)(s0 + ATT_D + 4);
            const float* f0 = (const float*)&r0a;
            const float* g0 = (const float*)&r0b;
            const float* f1 = (const float*)&r1a;
            const float* g1 = (const float*)&r1b;
            #pragma unroll
            for (int i = 0; i < 4; ++i) {
                union { _Float16 h[2]; unsigned u; } pk;
                pk.h[0] = (_Float16)f0[i];
                pk.h[1] = (_Float16)f1[i];
                *(unsigned*)&VT[(vcol0 + i) * VS + vrow] = pk.u;
                pk.h[0] = (_Float16)g0[i];
                pk.h[1] = (_Float16)g1[i];
                *(unsigned*)&VT[(vcol0 + 4 + i) * VS + vrow] = pk.u;
            }
        }
        // prefetch next tile while this one is consumed
        if (kt + BN <= kmax) {
            __builtin_prefetch(Kh + (size_t)(kt + BN + krow) * ATT_D + kcol0, 0, 0);
            __builtin_prefetch(Vh + (size_t)(kt + BN + vrow) * ATT_D + vcol0, 0, 0);
        }
        __syncthreads();                                   // tiles visible to all waves

        // ===== wave-uniform skip: tiles entirely above this wave's rows do nothing =====
        if (kt <= qhi) {
            // ============ scores: S = (Q/sqrt(d)) @ K^T , 16 x 64 (8 WMMAs) ============
            v8f c[4];
            #pragma unroll
            for (int t = 0; t < 4; ++t) {
                const _Float16* kr = &Kld[(16 * t + n) * KS + 16 * half];
                v16h b0 = ld_frag(kr);        // d 0..31
                v16h b1 = ld_frag(kr + 32);   // d 32..63
                v8f acc = (v8f){};
                acc = WMMA_F16(aQ0, b0, acc);
                acc = WMMA_F16(aQ1, b1, acc);
                c[t] = acc;
            }

            // ====== causal mask only on diagonal-crossing tiles (wave-uniform test) ======
            if (kt + (BN - 1) > qlo) {
                #pragma unroll
                for (int t = 0; t < 4; ++t) {
                    const int kcol = kt + 16 * t + n;
                    #pragma unroll
                    for (int r = 0; r < 8; ++r) {
                        const int qrow = qlo + r + 8 * half;
                        c[t][r] = (kcol > qrow) ? -1e30f : c[t][r];
                    }
                }
            }

            // ====== online softmax: row max via 4 bpermutes, exp, rescale ======
            float alphav[8];
            #pragma unroll
            for (int r = 0; r < 8; ++r) {
                float tm = fmaxf(fmaxf(c[0][r], c[1][r]), fmaxf(c[2][r], c[3][r]));
                tm = fmaxf(tm, __shfl_xor(tm, 1));
                tm = fmaxf(tm, __shfl_xor(tm, 2));
                tm = fmaxf(tm, __shfl_xor(tm, 4));
                tm = fmaxf(tm, __shfl_xor(tm, 8));
                const float mnew = fmaxf(mrun[r], tm);
                alphav[r] = __expf(mrun[r] - mnew);
                mrun[r] = mnew;
                c[0][r] = __expf(c[0][r] - mnew);
                c[1][r] = __expf(c[1][r] - mnew);
                c[2][r] = __expf(c[2][r] - mnew);
                c[3][r] = __expf(c[3][r] - mnew);
                oAcc[0][r] *= alphav[r];
                oAcc[1][r] *= alphav[r];
                oAcc[2][r] *= alphav[r];
                oAcc[3][r] *= alphav[r];
            }

            // ====== relayout P: C-fragment(f32) -> LDS -> A-fragments(f16) ======
            {
                _Float16* pw = &Pld[w][0];
                #pragma unroll
                for (int t = 0; t < 4; ++t)
                    #pragma unroll
                    for (int r = 0; r < 8; ++r)
                        pw[(r + 8 * half) * PS + 16 * t + n] = (_Float16)c[t][r];
            }
            asm volatile("s_wait_dscnt 0x0" ::: "memory"); // per-wave LDS round trip
            __builtin_amdgcn_wave_barrier();

            v16h aP0, aP1;                                 // k 0..31 / 32..63
            {
                const _Float16* pr = &Pld[w][n * PS];      // row m = lane & 15
                #pragma unroll
                for (int j = 0; j < 8; ++j) {
                    int k = half * 16 + (j << 1);
                    aP0[2*j]   = pr[k];
                    aP0[2*j+1] = pr[k + 1];
                    aP1[2*j]   = pr[32 + k];
                    aP1[2*j+1] = pr[32 + k + 1];
                }
            }
            __builtin_amdgcn_wave_barrier();

            // ====== row sums via WMMA: rs = P @ ones (2 WMMAs, lands in C layout) ======
            {
                v8f rs = (v8f){};
                rs = WMMA_F16(aP0, ones, rs);
                rs = WMMA_F16(aP1, ones, rs);
                #pragma unroll
                for (int r = 0; r < 8; ++r)
                    lrun[r] = lrun[r] * alphav[r] + rs[r];
            }

            // ====== O += P @ V (K=64, four 16-wide d tiles, 8 WMMAs) ======
            #pragma unroll
            for (int t2 = 0; t2 < 4; ++t2) {
                const _Float16* vr = &VT[(16 * t2 + n) * VS + 16 * half];
                v16h bv0 = ld_frag(vr);         // k 0..31
                v16h bv1 = ld_frag(vr + 32);    // k 32..63
                oAcc[t2] = WMMA_F16(aP0, bv0, oAcc[t2]);
                oAcc[t2] = WMMA_F16(aP1, bv1, oAcc[t2]);
            }
        }
    }

    // ================= normalize and store =================
    #pragma unroll
    for (int r = 0; r < 8; ++r) {
        const float inv = 1.0f / lrun[r];
        const int row = qlo + r + 8 * half;
        float* orow = Oh + (size_t)row * ATT_D + n;
        orow[0]  = oAcc[0][r] * inv;
        orow[16] = oAcc[1][r] * inv;
        orow[32] = oAcc[2][r] * inv;
        orow[48] = oAcc[3][r] * inv;
    }
}

extern "C" void kernel_launch(void* const* d_in, const int* in_sizes, int n_in,
                              void* d_out, int out_size, void* d_ws, size_t ws_size,
                              hipStream_t stream) {
    (void)in_sizes; (void)n_in; (void)out_size; (void)d_ws; (void)ws_size;
    const float* q = (const float*)d_in[0];
    const float* k = (const float*)d_in[1];
    const float* v = (const float*)d_in[2];
    float* out = (float*)d_out;

    dim3 grid(ATT_S / BM, ATT_B * ATT_H);   // 16 x 32 blocks
    dim3 block(256);                        // 8 waves of 32
    hipLaunchKernelGGL(fa2_causal_wmma_kernel, grid, block, 0, stream, q, k, v, out);
}
